// ExpertRouter_59957743452643
// MI455X (gfx1250) — matrix operand order repaired and checked by reading, not compiled, not run on hardware
//
#include <hip/hip_runtime.h>
#include <hip/hip_bf16.h>
#include <math.h>

// Problem constants (from reference)
#define T_TOK   4096
#define D_DIM   6144
#define MID_DIM 512
#define NE_EXP  16

// Tiling
#define BM 64                  // tokens per block
// f32 fallback path: 64x64 block tile
#define BNF 64
#define BKF 64
#define KPADF 4
#define NSTAGES_F (D_DIM / BKF)    // 96
// bf16 fast path: 64x128 block tile, wave tile 32x64 (2x4 WMMA)
#define BNB 128
#define BK2 128
#define KPAD2 8                     // bf16 pad -> row stride 272B (16B aligned)
#define NSTAGES_B (D_DIM / BK2)    // 48

typedef __attribute__((ext_vector_type(16))) __bf16 v16bf;
typedef __attribute__((ext_vector_type(4)))  __bf16 v4bf;
typedef __attribute__((ext_vector_type(8)))  float  v8f;
typedef __attribute__((vector_size(16)))     int    i128v;   // b128 payload

// ---------------------------------------------------------------------------
// CDNA5 async global->LDS copy (ASYNCcnt-tracked), with portable fallback.
// Signature (verified via compiler diagnostic): (v4i AS1*, v4i AS3*, Iint, Iint)
// ---------------------------------------------------------------------------
#if __has_builtin(__builtin_amdgcn_global_load_async_to_lds_b128) && \
    __has_builtin(__builtin_amdgcn_s_wait_asynccnt)
#define USE_ASYNC 1
#else
#define USE_ASYNC 0
#endif

__device__ __forceinline__ void cp16_g2l(const void* g, void* l)
{
#if USE_ASYNC
    __builtin_amdgcn_global_load_async_to_lds_b128(
        (__attribute__((address_space(1))) i128v*)g,
        (__attribute__((address_space(3))) i128v*)l,
        0, 0);
#else
    *(i128v*)l = *(const i128v*)g;
#endif
}

#if USE_ASYNC
#define WAIT_ASYNC(n) __builtin_amdgcn_s_wait_asynccnt(n)
#else
#define WAIT_ASYNC(n)
#endif

// ---------------------------------------------------------------------------
// Shared epilogue (templated on #N-tiles per wave): bias + relu + w2
// weighting, column reduce, atomic accumulate.
// C/D layout: VGPR v, half h -> row v+8h; lane&15 -> col.
// ---------------------------------------------------------------------------
template <int NI>
__device__ __forceinline__ void epilogue_t(v8f acc[2][NI],
                                           const float* __restrict__ b1,
                                           const float* __restrict__ w2,
                                           float* __restrict__ e_out,
                                           int rowbase, int colbase, int ex,
                                           int h, int l15)
{
    float b1v[NI], w2v[NI];
    #pragma unroll
    for (int ni = 0; ni < NI; ++ni) {
        const int c = colbase + ni * 16 + l15;
        b1v[ni] = b1[ex * MID_DIM + c];
        w2v[ni] = w2[ex * MID_DIM + c];
    }
    #pragma unroll
    for (int mi = 0; mi < 2; ++mi) {
        #pragma unroll
        for (int v = 0; v < 8; ++v) {
            float s = 0.0f;
            #pragma unroll
            for (int ni = 0; ni < NI; ++ni) {
                float hv = acc[mi][ni][v] + b1v[ni];
                hv = hv > 0.0f ? hv : 0.0f;
                s += hv * w2v[ni];
            }
            s += __shfl_xor(s, 8);
            s += __shfl_xor(s, 4);
            s += __shfl_xor(s, 2);
            s += __shfl_xor(s, 1);
            if (l15 == 0) {
                const int row = rowbase + mi * 16 + v + 8 * h;
                atomicAdd(&e_out[(size_t)row * NE_EXP + ex], s);
            }
        }
    }
}

// ===========================================================================
// FAST PATH: pre-converted bf16 operands, contiguous b128 fragment loads.
// xb  : bf16 [T][D]          (A, row-major in k)
// w1t : bf16 [NE][MID][D]    (B, TRANSPOSED: row-major in k)
// ===========================================================================

// --- pre-pass 1: x f32 -> bf16 -------------------------------------------
__global__ __launch_bounds__(256)
void cvt_x_kernel(const float* __restrict__ x, __bf16* __restrict__ xb)
{
    const size_t i = ((size_t)blockIdx.x * 256 + threadIdx.x) * 4;
    const float4 v = *(const float4*)(x + i);
    v4bf o;
    o[0] = (__bf16)v.x; o[1] = (__bf16)v.y;
    o[2] = (__bf16)v.z; o[3] = (__bf16)v.w;
    *(v4bf*)(xb + i) = o;
}

// --- pre-pass 2: w1 [n][k][m] f32 -> w1t [n][m][k] bf16 (LDS transpose) ---
__global__ __launch_bounds__(256)
void tr_w1_kernel(const float* __restrict__ w1, __bf16* __restrict__ w1t)
{
    __shared__ float tile[32][33];
    const int tid = threadIdx.x;
    const int n  = blockIdx.z;
    const int k0 = blockIdx.x * 32;
    const int m0 = blockIdx.y * 32;

    #pragma unroll
    for (int it = 0; it < 4; ++it) {
        const int kk = (tid >> 5) + it * 8;
        const int mm = tid & 31;
        tile[kk][mm] = w1[((size_t)n * D_DIM + k0 + kk) * MID_DIM + m0 + mm];
    }
    __syncthreads();
    #pragma unroll
    for (int it = 0; it < 4; ++it) {
        const int mm = (tid >> 5) + it * 8;
        const int kk = tid & 31;
        w1t[((size_t)n * MID_DIM + m0 + mm) * D_DIM + k0 + kk] = (__bf16)tile[kk][mm];
    }
}

// --- bf16 tile staging: 24 x 16B async chunks per thread per stage --------
__device__ __forceinline__ void stage_tiles_bf16(const __bf16* __restrict__ xb,
                                                 const __bf16* __restrict__ w1tbase,
                                                 int m0, int k0,
                                                 __bf16 (*lx)[BK2 + KPAD2],
                                                 __bf16 (*lw)[BK2 + KPAD2],
                                                 int tid)
{
    const int r0 = tid >> 4;            // 0..7
    const int c  = (tid & 15) << 3;     // chunk of 8 bf16 (16B): 0,8,...,120
    #pragma unroll
    for (int rr = 0; rr < 8; ++rr) {    // x: 64 rows
        const int r = r0 + rr * 8;
        cp16_g2l(xb + (size_t)(m0 + r) * D_DIM + k0 + c, &lx[r][c]);
    }
    #pragma unroll
    for (int rr = 0; rr < 16; ++rr) {   // w: 128 rows
        const int m = r0 + rr * 8;
        cp16_g2l(w1tbase + (size_t)m * D_DIM + k0 + c, &lw[m][c]);
    }
}

// --- 4 WMMA k-steps per stage, 2x4 wave tile: 12 b128 loads per 8 WMMAs ---
__device__ __forceinline__ void compute_stage_bf16(const __bf16 (*lx)[BK2 + KPAD2],
                                                   const __bf16 (*lw)[BK2 + KPAD2],
                                                   v8f acc[2][4],
                                                   int wm, int wn, int h, int l15)
{
    #pragma unroll
    for (int ks = 0; ks < BK2; ks += 32) {
        v16bf a[2], b[4];
        // A 16x32 bf16: lane l15=row; elem i<8 -> K=i+8h, i>=8 -> K=i+8+8h
        #pragma unroll
        for (int mi = 0; mi < 2; ++mi) {
            const int r = wm * 32 + mi * 16 + l15;
            const __bf16* p = &lx[r][ks + 8 * h];
            const __bf16* q = &lx[r][ks + 16 + 8 * h];
            #pragma unroll
            for (int i = 0; i < 8; ++i) {
                a[mi][i]     = p[i];
                a[mi][i + 8] = q[i];
            }
        }
        // B 32x16 bf16: lane l15=col; elem i<8 -> K=i+8h, i>=8 -> K=i+8+8h
        #pragma unroll
        for (int ni = 0; ni < 4; ++ni) {
            const int c = wn * 64 + ni * 16 + l15;
            const __bf16* p = &lw[c][ks + 8 * h];
            const __bf16* q = &lw[c][ks + 16 + 8 * h];
            #pragma unroll
            for (int i = 0; i < 8; ++i) {
                b[ni][i]     = p[i];
                b[ni][i + 8] = q[i];
            }
        }
        #pragma unroll
        for (int mi = 0; mi < 2; ++mi)
            #pragma unroll
            for (int ni = 0; ni < 4; ++ni)
                acc[mi][ni] = __builtin_amdgcn_wmma_f32_16x16x32_bf16(
                    false, a[mi], false, b[ni],
                    (short)0, acc[mi][ni], false, false);
    }
}

__global__ __launch_bounds__(128)
void expert_gemm_bf16_kernel(const __bf16* __restrict__ xb,
                             const __bf16* __restrict__ w1t,
                             const float* __restrict__ b1,
                             const float* __restrict__ w2,
                             float* __restrict__ e_out)
{
    __shared__ __bf16 lds_x[2][BM][BK2 + KPAD2];    // 2 x 17 KB
    __shared__ __bf16 lds_w[2][BNB][BK2 + KPAD2];   // 2 x 34 KB

    const int m0 = blockIdx.x * BM;
    const int n0 = blockIdx.y * BNB;
    const int ex = blockIdx.z;

    const int tid  = threadIdx.x;
    const int wid  = tid >> 5;
    const int lane = tid & 31;
    const int h    = lane >> 4;
    const int l15  = lane & 15;
    const int wm   = wid >> 1;
    const int wn   = wid & 1;

    v8f acc[2][4];
    #pragma unroll
    for (int mi = 0; mi < 2; ++mi)
        #pragma unroll
        for (int ni = 0; ni < 4; ++ni)
            #pragma unroll
            for (int v = 0; v < 8; ++v)
                acc[mi][ni][v] = 0.0f;

    const __bf16* w1tbase = w1t + ((size_t)ex * MID_DIM + n0) * D_DIM;

    stage_tiles_bf16(xb, w1tbase, m0, 0, lds_x[0], lds_w[0], tid);

    for (int s = 0; s < NSTAGES_B; ++s) {
        const int cur = s & 1;
        if (s + 1 < NSTAGES_B) {
            stage_tiles_bf16(xb, w1tbase, m0, (s + 1) * BK2,
                             lds_x[cur ^ 1], lds_w[cur ^ 1], tid);
            WAIT_ASYNC(24);    // 24 newest = next stage's copies
        } else {
            WAIT_ASYNC(0);
        }
        __syncthreads();
        compute_stage_bf16(lds_x[cur], lds_w[cur], acc, wm, wn, h, l15);
        __syncthreads();
    }

    epilogue_t<4>(acc, b1, w2, e_out,
                  m0 + wm * 32, n0 + wn * 64, ex, h, l15);
}

// ===========================================================================
// FALLBACK PATH (small workspace): f32 tiles in LDS, cvt in fragment build.
// ===========================================================================
__device__ __forceinline__ void stage_tiles_f32(const float* __restrict__ x,
                                                const float* __restrict__ w1base,
                                                int m0, int k0,
                                                float (*lx)[BKF + KPADF],
                                                float (*lw)[BNF + KPADF],
                                                int tid)
{
    const int r0 = tid >> 4;
    const int c  = (tid & 15) << 2;
    #pragma unroll
    for (int rr = 0; rr < 8; ++rr) {
        const int r = r0 + rr * 8;
        cp16_g2l(x + (size_t)(m0 + r) * D_DIM + k0 + c, &lx[r][c]);
    }
    #pragma unroll
    for (int rr = 0; rr < 8; ++rr) {
        const int k = r0 + rr * 8;
        cp16_g2l(w1base + (size_t)(k0 + k) * MID_DIM + c, &lw[k][c]);
    }
}

__device__ __forceinline__ void compute_stage_f32(const float (*lx)[BKF + KPADF],
                                                  const float (*lw)[BNF + KPADF],
                                                  v8f acc[2][2],
                                                  int wm, int wn, int h, int l15)
{
    #pragma unroll
    for (int ks = 0; ks < BKF; ks += 32) {
        v16bf a[2], b[2];
        #pragma unroll
        for (int mi = 0; mi < 2; ++mi) {
            const int r = wm * 32 + mi * 16 + l15;
            const float* p0 = &lx[r][ks + 8 * h];
            const float* p1 = &lx[r][ks + 16 + 8 * h];
            #pragma unroll
            for (int i = 0; i < 8; ++i) {
                a[mi][i]     = (__bf16)p0[i];
                a[mi][i + 8] = (__bf16)p1[i];
            }
        }
        #pragma unroll
        for (int ni = 0; ni < 2; ++ni) {
            const int c = wn * 32 + ni * 16 + l15;
            #pragma unroll
            for (int i = 0; i < 16; ++i)
                b[ni][i] = (__bf16)lw[ks + (i & 7) + 8 * h + 16 * (i >> 3)][c];
        }
        #pragma unroll
        for (int mi = 0; mi < 2; ++mi)
            #pragma unroll
            for (int ni = 0; ni < 2; ++ni)
                acc[mi][ni] = __builtin_amdgcn_wmma_f32_16x16x32_bf16(
                    false, a[mi], false, b[ni],
                    (short)0, acc[mi][ni], false, false);
    }
}

__global__ __launch_bounds__(128)
void expert_gemm_f32_kernel(const float* __restrict__ x,
                            const float* __restrict__ w1,
                            const float* __restrict__ b1,
                            const float* __restrict__ w2,
                            float* __restrict__ e_out)
{
    __shared__ float lds_x[2][BM][BKF + KPADF];
    __shared__ float lds_w[2][BKF][BNF + KPADF];

    const int m0 = blockIdx.x * BM;
    const int n0 = blockIdx.y * BNF;
    const int ex = blockIdx.z;

    const int tid  = threadIdx.x;
    const int wid  = tid >> 5;
    const int lane = tid & 31;
    const int h    = lane >> 4;
    const int l15  = lane & 15;
    const int wm   = wid >> 1;
    const int wn   = wid & 1;

    v8f acc[2][2];
    #pragma unroll
    for (int mi = 0; mi < 2; ++mi)
        #pragma unroll
        for (int ni = 0; ni < 2; ++ni)
            #pragma unroll
            for (int v = 0; v < 8; ++v)
                acc[mi][ni][v] = 0.0f;

    const float* w1base = w1 + (size_t)ex * D_DIM * MID_DIM + n0;

    stage_tiles_f32(x, w1base, m0, 0, lds_x[0], lds_w[0], tid);

    for (int s = 0; s < NSTAGES_F; ++s) {
        const int cur = s & 1;
        if (s + 1 < NSTAGES_F) {
            stage_tiles_f32(x, w1base, m0, (s + 1) * BKF,
                            lds_x[cur ^ 1], lds_w[cur ^ 1], tid);
            WAIT_ASYNC(16);
        } else {
            WAIT_ASYNC(0);
        }
        __syncthreads();
        compute_stage_f32(lds_x[cur], lds_w[cur], acc, wm, wn, h, l15);
        __syncthreads();
    }

    epilogue_t<2>(acc, b1, w2, e_out,
                  m0 + wm * 32, n0 + wn * 32, ex, h, l15);
}

// ---------------------------------------------------------------------------
// Kernel: gating softmax + combine + sigmoid. One wave per token.
// ---------------------------------------------------------------------------
__global__ __launch_bounds__(256)
void gate_combine_kernel(const float* __restrict__ x,
                         const float* __restrict__ gw,
                         const float* __restrict__ gb,
                         const float* __restrict__ b2,
                         const float* __restrict__ e_in,
                         float* __restrict__ out)
{
    const int wid  = threadIdx.x >> 5;
    const int lane = threadIdx.x & 31;
    const int t    = blockIdx.x * 8 + wid;

    float acc[NE_EXP];
    #pragma unroll
    for (int n = 0; n < NE_EXP; ++n) acc[n] = 0.0f;

    const float* xr = x + (size_t)t * D_DIM;
    for (int k = lane; k < D_DIM; k += 32) {
        const float xv = xr[k];
        const float4* g4 = (const float4*)(gw + (size_t)k * NE_EXP);
        const float4 g0 = g4[0], g1 = g4[1], g2 = g4[2], g3 = g4[3];
        acc[0]  += xv * g0.x;  acc[1]  += xv * g0.y;
        acc[2]  += xv * g0.z;  acc[3]  += xv * g0.w;
        acc[4]  += xv * g1.x;  acc[5]  += xv * g1.y;
        acc[6]  += xv * g1.z;  acc[7]  += xv * g1.w;
        acc[8]  += xv * g2.x;  acc[9]  += xv * g2.y;
        acc[10] += xv * g2.z;  acc[11] += xv * g2.w;
        acc[12] += xv * g3.x;  acc[13] += xv * g3.y;
        acc[14] += xv * g3.z;  acc[15] += xv * g3.w;
    }
    #pragma unroll
    for (int n = 0; n < NE_EXP; ++n) {
        #pragma unroll
        for (int off = 16; off >= 1; off >>= 1)
            acc[n] += __shfl_xor(acc[n], off);
    }

    float logit[NE_EXP];
    float mx = -1e30f;
    #pragma unroll
    for (int n = 0; n < NE_EXP; ++n) {
        logit[n] = acc[n] + gb[n];
        mx = fmaxf(mx, logit[n]);
    }
    float se = 0.0f;
    #pragma unroll
    for (int n = 0; n < NE_EXP; ++n) {
        logit[n] = __expf(logit[n] - mx);
        se += logit[n];
    }
    float sv = 0.0f;
    #pragma unroll
    for (int n = 0; n < NE_EXP; ++n)
        sv += (logit[n] / se) * (e_in[(size_t)t * NE_EXP + n] + b2[n]);

    if (lane == 0)
        out[t] = 1.0f / (1.0f + __expf(-sv));
}

// ---------------------------------------------------------------------------
extern "C" void kernel_launch(void* const* d_in, const int* in_sizes, int n_in,
                              void* d_out, int out_size, void* d_ws, size_t ws_size,
                              hipStream_t stream)
{
    const float* x  = (const float*)d_in[0];   // [4096, 6144]
    const float* w1 = (const float*)d_in[1];   // [16, 6144, 512]
    const float* b1 = (const float*)d_in[2];   // [16, 512]
    const float* w2 = (const float*)d_in[3];   // [16, 512]
    const float* b2 = (const float*)d_in[4];   // [16]
    const float* gw = (const float*)d_in[5];   // [6144, 16]
    const float* gb = (const float*)d_in[6];   // [16]

    // workspace layout
    const size_t e_bytes   = (size_t)T_TOK * NE_EXP * sizeof(float);      // 256 KB
    const size_t off_xb    = e_bytes;
    const size_t xb_bytes  = (size_t)T_TOK * D_DIM * 2;                   // 48 MB
    const size_t off_w1t   = off_xb + xb_bytes;
    const size_t w1t_bytes = (size_t)NE_EXP * MID_DIM * D_DIM * 2;        // 96 MB
    const size_t needed    = off_w1t + w1t_bytes;

    float* e = (float*)d_ws;
    (void)hipMemsetAsync(e, 0, e_bytes, stream);

    if (ws_size >= needed) {
        __bf16* xb  = (__bf16*)((char*)d_ws + off_xb);
        __bf16* w1t = (__bf16*)((char*)d_ws + off_w1t);

        cvt_x_kernel<<<dim3((T_TOK * (size_t)D_DIM) / (4 * 256)), dim3(256),
                       0, stream>>>(x, xb);
        tr_w1_kernel<<<dim3(D_DIM / 32, MID_DIM / 32, NE_EXP), dim3(256),
                       0, stream>>>(w1, w1t);
        expert_gemm_bf16_kernel<<<dim3(T_TOK / BM, MID_DIM / BNB, NE_EXP),
                                  dim3(128), 0, stream>>>(xb, w1t, b1, w2, e);
    } else {
        expert_gemm_f32_kernel<<<dim3(T_TOK / BM, MID_DIM / BNF, NE_EXP),
                                 dim3(128), 0, stream>>>(x, w1, b1, w2, e);
    }

    gate_combine_kernel<<<dim3(T_TOK / 8), dim3(256), 0, stream>>>(
        x, gw, gb, b2, e, (float*)d_out);
}
